// SparseMoERouter_87875030876694
// MI455X (gfx1250) — matrix operand order repaired and checked by reading, compiled, not run on hardware
//
#include <hip/hip_runtime.h>
#include <hip/hip_bf16.h>

typedef __bf16 bf16_t;
typedef __attribute__((ext_vector_type(4)))  __bf16 bvec4;
typedef __attribute__((ext_vector_type(8)))  __bf16 bvec8;
typedef __attribute__((ext_vector_type(16))) __bf16 bvec16;
typedef __attribute__((ext_vector_type(8)))  float  fvec8;

#define DIM_   1024
#define HID_   4096
#define NE_    8
#define KTOP_  2
#define NTOK_  32768
#define CAP_   10240          // ceil(1.25 * 32768 * 2 / 8)
#define NBINS_ 1024

// ---------------- workspace layout (bytes) ----------------
#define OFF_TOPI   ((size_t)0)                         // int[NTOK_*KTOP_]
#define OFF_TOPW   ((size_t)262144)                    // float[NTOK_*KTOP_]
#define OFF_HIST   ((size_t)524288)                    // int[NE_*NBINS_]
#define OFF_POS    ((size_t)557056)                    // int[8]
#define OFF_TIE    ((size_t)557088)                    // int[8]   (adjacent to POS)
#define OFF_THR    ((size_t)557120)                    // int[8]
#define OFF_REM    ((size_t)557152)                    // int[8]
#define OFF_USAGE  ((size_t)557184)                    // float[8]
#define OFF_ZSUM   ((size_t)557216)                    // float[1]
#define OFF_TOKL   ((size_t)1048576)                   // int[NE_*CAP_]
#define OFF_GWL    ((size_t)1376256)                   // float[NE_*CAP_]
#define OFF_W1T    ((size_t)2097152)                   // bf16[NE_*HID_*DIM_]
#define OFF_W2T    (OFF_W1T + (size_t)NE_*HID_*DIM_*2) // bf16[NE_*DIM_*HID_]
#define OFF_H      (OFF_W2T + (size_t)NE_*DIM_*HID_*2) // bf16[CAP_*HID_]
#define OFF_ABUF   (OFF_H   + (size_t)CAP_*HID_*2)     // bf16[CAP_*DIM_]

// ---------------- fragment helpers ----------------
// A (16x32 bf16) per-lane layout: rows m=lane%16; K = 16*half + [0..15] contiguous
__device__ __forceinline__ bvec16 a_frag_bf16(const bf16_t* p) {
  bvec8 c0 = *(const bvec8*)(p);
  bvec8 c1 = *(const bvec8*)(p + 8);
  return __builtin_shufflevector(c0, c1, 0,1,2,3,4,5,6,7,8,9,10,11,12,13,14,15);
}
// B (32x16 bf16) per-lane layout: col n=lane%16; K = 16*half + [0..15] contiguous
__device__ __forceinline__ bvec16 b_frag(const bf16_t* p) { return *(const bvec16*)(p); }

__device__ __forceinline__ float gelu_exact(float v) {
  return 0.5f * v * (1.0f + erff(v * 0.70710678118654752f));
}

// ---------------- init ----------------
__global__ void moe_init(float* out, int* tokL, float* gwL, int* hist,
                         int* ctrs, float* usage, float* zsum) {
  size_t stride = (size_t)gridDim.x * blockDim.x;
  size_t i0 = (size_t)blockIdx.x * blockDim.x + threadIdx.x;
  const size_t nOut = (size_t)NTOK_ * DIM_ + 1;
  for (size_t i = i0; i < nOut; i += stride) out[i] = 0.0f;
  for (size_t i = i0; i < (size_t)NE_ * CAP_; i += stride) { tokL[i] = 0; gwL[i] = 0.0f; }
  for (size_t i = i0; i < (size_t)NE_ * NBINS_; i += stride) hist[i] = 0;
  if (i0 < 16) ctrs[i0] = 0;                // pos[8] + tie[8]
  if (i0 < NE_) usage[i0] = 0.0f;
  if (i0 == 0) zsum[0] = 0.0f;
}

// ---------------- transpose + fp32->bf16 convert: src[e][Kd][Nd] -> dst[e][Nd][Kd] ----------------
__global__ __launch_bounds__(256) void moe_cvt_transpose(const float* __restrict__ src,
                                                         bf16_t* __restrict__ dst,
                                                         int Kd, int Nd) {
  __shared__ float tile[32][33];
  int e = blockIdx.z;
  const float* s = src + (size_t)e * Kd * Nd;
  bf16_t* d = dst + (size_t)e * Kd * Nd;
  int n0 = blockIdx.x * 32, k0 = blockIdx.y * 32;
  int tx = threadIdx.x, ty = threadIdx.y;
  for (int i = ty; i < 32; i += 8)
    tile[i][tx] = s[(size_t)(k0 + i) * Nd + (n0 + tx)];
  __syncthreads();
  for (int i = ty; i < 32; i += 8)
    d[(size_t)(n0 + i) * Kd + (k0 + tx)] = (bf16_t)tile[tx][i];
}

// ---------------- router ----------------
__global__ __launch_bounds__(256) void moe_router(const float* __restrict__ x,
                                                  const float* __restrict__ rw,
                                                  int* __restrict__ topi,
                                                  float* __restrict__ topw,
                                                  int* __restrict__ hist,
                                                  float* __restrict__ usage,
                                                  float* __restrict__ zsum) {
  __shared__ float s_rw[NE_ * DIM_];
  __shared__ float s_usage[NE_];
  __shared__ float s_z;
  for (int i = threadIdx.x; i < NE_ * DIM_; i += 256) s_rw[i] = rw[i];
  if (threadIdx.x < NE_) s_usage[threadIdx.x] = 0.0f;
  if (threadIdx.x == 0) s_z = 0.0f;
  __syncthreads();

  int wv = threadIdx.x >> 5, lane = threadIdx.x & 31;
  int t = blockIdx.x * 8 + wv;
  const float* xr = x + (size_t)t * DIM_;
  float acc[NE_] = {};
  for (int k = lane; k < DIM_; k += 32) {
    float xv = xr[k];
#pragma unroll
    for (int e = 0; e < NE_; ++e) acc[e] += xv * s_rw[e * DIM_ + k];
  }
#pragma unroll
  for (int e = 0; e < NE_; ++e) {
#pragma unroll
    for (int off = 16; off > 0; off >>= 1) acc[e] += __shfl_xor(acc[e], off, 32);
  }
  if (lane == 0) {
    float m = acc[0];
#pragma unroll
    for (int e = 1; e < NE_; ++e) m = fmaxf(m, acc[e]);
    float p[NE_], s = 0.0f;
#pragma unroll
    for (int e = 0; e < NE_; ++e) { p[e] = expf(acc[e] - m); s += p[e]; }
    float inv = 1.0f / s;
#pragma unroll
    for (int e = 0; e < NE_; ++e) p[e] *= inv;
    float z = m + logf(s);
    int i1 = 0; float p1 = p[0];
#pragma unroll
    for (int e = 1; e < NE_; ++e) if (p[e] > p1) { p1 = p[e]; i1 = e; }
    int i2 = (i1 == 0) ? 1 : 0; float p2 = p[i2];
#pragma unroll
    for (int e = 0; e < NE_; ++e) if (e != i1 && p[e] > p2) { p2 = p[e]; i2 = e; }
    float sm = fmaxf(p1 + p2, 1e-9f);
    float wa = p1 / sm, wb = p2 / sm;
    topi[2 * t] = i1; topi[2 * t + 1] = i2;
    topw[2 * t] = wa; topw[2 * t + 1] = wb;
    int b1i = min(NBINS_ - 1, (int)(wa * (float)NBINS_));
    int b2i = min(NBINS_ - 1, (int)(wb * (float)NBINS_));
    atomicAdd(&hist[i1 * NBINS_ + b1i], 1);
    atomicAdd(&hist[i2 * NBINS_ + b2i], 1);
#pragma unroll
    for (int e = 0; e < NE_; ++e) atomicAdd(&s_usage[e], p[e]);
    atomicAdd(&s_z, z * z);
  }
  __syncthreads();
  if (threadIdx.x < NE_) atomicAdd(&usage[threadIdx.x], s_usage[threadIdx.x]);
  if (threadIdx.x == 0) atomicAdd(zsum, s_z);
}

// ---------------- capacity threshold ----------------
__global__ void moe_thresh(const int* __restrict__ hist, int* thr, int* rem) {
  int e = threadIdx.x;
  if (e >= NE_) return;
  const int* hE = hist + e * NBINS_;
  int total = 0;
  for (int b = 0; b < NBINS_; ++b) total += hE[b];
  if (total <= CAP_) { thr[e] = -1; rem[e] = 0; return; }
  int cum = 0;
  for (int b = NBINS_ - 1; b >= 0; --b) {
    cum += hE[b];
    if (cum >= CAP_) { thr[e] = b; rem[e] = CAP_ - (cum - hE[b]); return; }
  }
}

// ---------------- build kept-slot lists ----------------
__global__ void moe_build(const int* __restrict__ topi, const float* __restrict__ topw,
                          const int* __restrict__ thr, const int* __restrict__ rem,
                          int* tie, int* pos, int* __restrict__ tokL, float* __restrict__ gwL) {
  int s = blockIdx.x * blockDim.x + threadIdx.x;
  if (s >= NTOK_ * KTOP_) return;
  int e = topi[s];
  float w = topw[s];
  int b = min(NBINS_ - 1, (int)(w * (float)NBINS_));
  int t = thr[e];
  bool accept = (b > t);
  if (!accept && b == t) accept = (atomicAdd(&tie[e], 1) < rem[e]);
  if (accept) {
    int p = atomicAdd(&pos[e], 1);
    if (p < CAP_) {
      tokL[e * CAP_ + p] = s >> 1;
      gwL[e * CAP_ + p]  = w;
    }
  }
}

// ---------------- gather + convert A rows for one expert ----------------
__global__ __launch_bounds__(256) void moe_gather(const float* __restrict__ x,
                                                  const int* __restrict__ tokL,
                                                  bf16_t* __restrict__ abuf, int e) {
  int r = blockIdx.x;                    // 0..CAP_-1
  int tok = tokL[e * CAP_ + r];
  const float4* src = (const float4*)(x + (size_t)tok * DIM_);
  bf16_t* dst = abuf + (size_t)r * DIM_;
  int i = threadIdx.x;                   // DIM_/4 == 256
  float4 f = src[i];
  bvec4 o;
  o[0] = (bf16_t)f.x; o[1] = (bf16_t)f.y; o[2] = (bf16_t)f.z; o[3] = (bf16_t)f.w;
  *(bvec4*)(dst + 4 * i) = o;
}

// ---------------- GEMM1: h = gelu(abuf @ W1 + b1) ----------------
__global__ __launch_bounds__(256) void moe_ffn1(const bf16_t* __restrict__ abuf,
                                                const bf16_t* __restrict__ w1t,
                                                const float* __restrict__ b1,
                                                bf16_t* __restrict__ h, int e) {
  const int lane = threadIdx.x & 31;
  const int wv = threadIdx.x >> 5;
  const int wm = wv >> 1, wn = wv & 1;
  const int half = lane >> 4, l16 = lane & 15;
  const int rowBase = blockIdx.y * 128 + wm * 32;
  const int colBase = blockIdx.x * 128 + wn * 64;

  const bf16_t* pa = abuf + (size_t)(rowBase + l16) * DIM_ + half * 16;
  const bf16_t* pb = w1t + (size_t)e * HID_ * DIM_
                   + (size_t)(colBase + l16) * DIM_ + half * 16;

  fvec8 acc[2][4] = {};
#pragma unroll 1
  for (int k0 = 0; k0 < DIM_; k0 += 32) {
    bvec16 a0 = a_frag_bf16(pa);
    bvec16 a1 = a_frag_bf16(pa + 16 * DIM_);
    bvec16 bf0 = b_frag(pb);
    bvec16 bf1 = b_frag(pb + 16 * DIM_);
    bvec16 bf2 = b_frag(pb + 32 * DIM_);
    bvec16 bf3 = b_frag(pb + 48 * DIM_);
    __builtin_prefetch((const void*)(pb + 512), 0, 1);   // 1KB lookahead on weight stream
    acc[0][0] = __builtin_amdgcn_wmma_f32_16x16x32_bf16(false, a0, false, bf0, (short)0, acc[0][0], false, false);
    acc[1][0] = __builtin_amdgcn_wmma_f32_16x16x32_bf16(false, a1, false, bf0, (short)0, acc[1][0], false, false);
    acc[0][1] = __builtin_amdgcn_wmma_f32_16x16x32_bf16(false, a0, false, bf1, (short)0, acc[0][1], false, false);
    acc[1][1] = __builtin_amdgcn_wmma_f32_16x16x32_bf16(false, a1, false, bf1, (short)0, acc[1][1], false, false);
    acc[0][2] = __builtin_amdgcn_wmma_f32_16x16x32_bf16(false, a0, false, bf2, (short)0, acc[0][2], false, false);
    acc[1][2] = __builtin_amdgcn_wmma_f32_16x16x32_bf16(false, a1, false, bf2, (short)0, acc[1][2], false, false);
    acc[0][3] = __builtin_amdgcn_wmma_f32_16x16x32_bf16(false, a0, false, bf3, (short)0, acc[0][3], false, false);
    acc[1][3] = __builtin_amdgcn_wmma_f32_16x16x32_bf16(false, a1, false, bf3, (short)0, acc[1][3], false, false);
    pa += 32; pb += 32;
  }
#pragma unroll
  for (int mi = 0; mi < 2; ++mi)
#pragma unroll
    for (int ni = 0; ni < 4; ++ni) {
      int n = colBase + ni * 16 + l16;
      float bias = b1[e * HID_ + n];
#pragma unroll
      for (int v = 0; v < 8; ++v) {
        int m = rowBase + mi * 16 + half * 8 + v;
        float val = acc[mi][ni][v] + bias;
        h[(size_t)m * HID_ + n] = (bf16_t)gelu_exact(val);
      }
    }
}

// ---------------- GEMM2: out[tok] += gw * (h @ W2 + b2) ----------------
__global__ __launch_bounds__(256) void moe_ffn2(const bf16_t* __restrict__ h,
                                                const bf16_t* __restrict__ w2t,
                                                const float* __restrict__ b2,
                                                const int* __restrict__ tokL,
                                                const float* __restrict__ gwL,
                                                float* __restrict__ out, int e) {
  const int lane = threadIdx.x & 31;
  const int wv = threadIdx.x >> 5;
  const int wm = wv >> 1, wn = wv & 1;
  const int half = lane >> 4, l16 = lane & 15;
  const int rowBase = blockIdx.y * 128 + wm * 32;
  const int colBase = blockIdx.x * 128 + wn * 64;

  const bf16_t* pa = h + (size_t)(rowBase + l16) * HID_ + half * 16;
  const bf16_t* pb = w2t + (size_t)e * DIM_ * HID_
                   + (size_t)(colBase + l16) * HID_ + half * 16;

  fvec8 acc[2][4] = {};
#pragma unroll 1
  for (int k0 = 0; k0 < HID_; k0 += 32) {
    bvec16 a0 = a_frag_bf16(pa);
    bvec16 a1 = a_frag_bf16(pa + 16 * HID_);
    bvec16 bf0 = b_frag(pb);
    bvec16 bf1 = b_frag(pb + 16 * HID_);
    bvec16 bf2 = b_frag(pb + 32 * HID_);
    bvec16 bf3 = b_frag(pb + 48 * HID_);
    __builtin_prefetch((const void*)(pb + 512), 0, 1);
    acc[0][0] = __builtin_amdgcn_wmma_f32_16x16x32_bf16(false, a0, false, bf0, (short)0, acc[0][0], false, false);
    acc[1][0] = __builtin_amdgcn_wmma_f32_16x16x32_bf16(false, a1, false, bf0, (short)0, acc[1][0], false, false);
    acc[0][1] = __builtin_amdgcn_wmma_f32_16x16x32_bf16(false, a0, false, bf1, (short)0, acc[0][1], false, false);
    acc[1][1] = __builtin_amdgcn_wmma_f32_16x16x32_bf16(false, a1, false, bf1, (short)0, acc[1][1], false, false);
    acc[0][2] = __builtin_amdgcn_wmma_f32_16x16x32_bf16(false, a0, false, bf2, (short)0, acc[0][2], false, false);
    acc[1][2] = __builtin_amdgcn_wmma_f32_16x16x32_bf16(false, a1, false, bf2, (short)0, acc[1][2], false, false);
    acc[0][3] = __builtin_amdgcn_wmma_f32_16x16x32_bf16(false, a0, false, bf3, (short)0, acc[0][3], false, false);
    acc[1][3] = __builtin_amdgcn_wmma_f32_16x16x32_bf16(false, a1, false, bf3, (short)0, acc[1][3], false, false);
    pa += 32; pb += 32;
  }
  const int* tl = tokL + e * CAP_;
  const float* gl = gwL + e * CAP_;
#pragma unroll
  for (int mi = 0; mi < 2; ++mi)
#pragma unroll
    for (int ni = 0; ni < 4; ++ni) {
      int n = colBase + ni * 16 + l16;
      float bias = b2[e * DIM_ + n];
#pragma unroll
      for (int v = 0; v < 8; ++v) {
        int m = rowBase + mi * 16 + half * 8 + v;
        float gw = gl[m];
        if (gw != 0.0f) {
          float* o = out + (size_t)tl[m] * DIM_ + n;
          *o += (acc[mi][ni][v] + bias) * gw;   // token unique within an expert -> no atomic
        }
      }
    }
}

// ---------------- aux loss ----------------
__global__ void moe_finalize(const int* __restrict__ pos, const float* __restrict__ usage,
                             const float* __restrict__ zsum, float* __restrict__ aux) {
  if (threadIdx.x != 0 || blockIdx.x != 0) return;
  float proc[NE_], psum = 0.0f;
#pragma unroll
  for (int e = 0; e < NE_; ++e) { proc[e] = (float)pos[e]; psum += proc[e]; }
  psum = fmaxf(psum, 1.0f);
  float lb = 0.0f;
#pragma unroll
  for (int e = 0; e < NE_; ++e)
    lb += (usage[e] / (float)NTOK_) * (proc[e] / psum);
  lb *= (float)NE_;
  float zl = zsum[0] / (float)NTOK_;
  aux[0] = 0.01f * lb + 0.01f * zl;
}

extern "C" void kernel_launch(void* const* d_in, const int* in_sizes, int n_in,
                              void* d_out, int out_size, void* d_ws, size_t ws_size,
                              hipStream_t stream) {
  (void)in_sizes; (void)n_in; (void)out_size; (void)ws_size;
  const float* x  = (const float*)d_in[0];
  const float* rw = (const float*)d_in[1];
  const float* w1 = (const float*)d_in[2];
  const float* b1 = (const float*)d_in[3];
  const float* w2 = (const float*)d_in[4];
  const float* b2 = (const float*)d_in[5];
  float* out = (float*)d_out;

  char* ws = (char*)d_ws;
  int*    topi  = (int*)   (ws + OFF_TOPI);
  float*  topw  = (float*) (ws + OFF_TOPW);
  int*    hist  = (int*)   (ws + OFF_HIST);
  int*    pos   = (int*)   (ws + OFF_POS);
  int*    tie   = (int*)   (ws + OFF_TIE);
  int*    thr   = (int*)   (ws + OFF_THR);
  int*    rem   = (int*)   (ws + OFF_REM);
  float*  usage = (float*) (ws + OFF_USAGE);
  float*  zsum  = (float*) (ws + OFF_ZSUM);
  int*    tokL  = (int*)   (ws + OFF_TOKL);
  float*  gwL   = (float*) (ws + OFF_GWL);
  bf16_t* w1t   = (bf16_t*)(ws + OFF_W1T);
  bf16_t* w2t   = (bf16_t*)(ws + OFF_W2T);
  bf16_t* hbuf  = (bf16_t*)(ws + OFF_H);
  bf16_t* abuf  = (bf16_t*)(ws + OFF_ABUF);

  moe_init<<<4096, 256, 0, stream>>>(out, tokL, gwL, hist, pos, usage, zsum);
  moe_cvt_transpose<<<dim3(HID_ / 32, DIM_ / 32, NE_), dim3(32, 8), 0, stream>>>(w1, w1t, DIM_, HID_);
  moe_cvt_transpose<<<dim3(DIM_ / 32, HID_ / 32, NE_), dim3(32, 8), 0, stream>>>(w2, w2t, HID_, DIM_);
  moe_router<<<NTOK_ / 8, 256, 0, stream>>>(x, rw, topi, topw, hist, usage, zsum);
  moe_thresh<<<1, 64, 0, stream>>>(hist, thr, rem);
  moe_build<<<(NTOK_ * KTOP_) / 256, 256, 0, stream>>>(topi, topw, thr, rem, tie, pos, tokL, gwL);
  for (int e = 0; e < NE_; ++e) {
    moe_gather<<<CAP_, 256, 0, stream>>>(x, tokL, abuf, e);
    moe_ffn1<<<dim3(HID_ / 128, CAP_ / 128), 256, 0, stream>>>(abuf, w1t, b1, hbuf, e);
    moe_ffn2<<<dim3(DIM_ / 128, CAP_ / 128), 256, 0, stream>>>(hbuf, w2t, b2, tokL, gwL, out, e);
  }
  moe_finalize<<<1, 32, 0, stream>>>(pos, usage, zsum, out + (size_t)NTOK_ * DIM_);
}
